// AttentionBlock_16174846836824
// MI455X (gfx1250) — matrix operand order repaired and checked by reading, not compiled, not run on hardware
//
#include <hip/hip_runtime.h>
#include <math.h>

// Problem constants (from reference): B=4, C=256, H=W=64 -> N=4096, RED=32
#define BATCH 4
#define CH    256
#define NPIX  4096
#define RED   32

typedef _Float16 f16;
typedef __attribute__((ext_vector_type(16))) _Float16 v16h;
typedef __attribute__((ext_vector_type(8)))  _Float16 v8h;
typedef __attribute__((ext_vector_type(8)))  float    v8f;

// ---------------------------------------------------------------------------
// WMMA wrapper: D = A(16x32 f16) * B(32x16 f16) + C(16x16 f32)
// ---------------------------------------------------------------------------
__device__ __forceinline__ v8f wmma16x16x32(v16h a, v16h b, v8f c) {
    return __builtin_amdgcn_wmma_f32_16x16x32_f16(
        /*neg_a=*/false, a, /*neg_b=*/false, b,
        /*c_mod=*/(short)0, c, /*reuse_a=*/false, /*reuse_b=*/false);
}

__device__ __forceinline__ v8f vzero8() {
    v8f z;
#pragma unroll
    for (int e = 0; e < 8; ++e) z[e] = 0.0f;
    return z;
}

// ---------------------------------------------------------------------------
// CDNA5 async global -> LDS copy (16B per lane), tracked on ASYNCcnt.
// VDST = wave-relative LDS byte address (low 32 bits of a generic LDS ptr),
// VADDR = 64-bit global address.  (CDNA5 ISA)
// ---------------------------------------------------------------------------
__device__ __forceinline__ void async_copy_b128(void* lds_ptr, const void* gptr) {
    unsigned int lds_addr = (unsigned int)(unsigned long long)(uintptr_t)lds_ptr;
    asm volatile("global_load_async_to_lds_b128 %0, %1, off"
                 :
                 : "v"(lds_addr), "v"((unsigned long long)(uintptr_t)gptr)
                 : "memory");
}
__device__ __forceinline__ void wait_asynccnt0() {
    asm volatile("s_wait_asynccnt 0" ::: "memory");
}

// ---------------------------------------------------------------------------
// A-matrix 16x32 f16 load from row-major [rows x ld] storage.
// Layout: lane r (0..15) + half h: VGPR0..3 hold K = h*16 + 0..7,
// VGPR4..7 hold K = h*16 + 8..15 (matches B).  => two contiguous 16B loads per lane.
// ---------------------------------------------------------------------------
__device__ __forceinline__ v16h load_tileA_f16(const f16* src, int row0, int col0,
                                               int ld, int r, int hf) {
    const f16* p = src + (size_t)(row0 + r) * ld + col0 + (hf << 4);
    v8h lo = *(const v8h*)p;
    v8h hi = *(const v8h*)(p + 8);
    v16h a;
#pragma unroll
    for (int e = 0; e < 8; ++e) { a[e] = lo[e]; a[e + 8] = hi[e]; }
    return a;
}

// ---------------------------------------------------------------------------
// B-matrix 32x16 f16 load. Layout used here (same as A): lane = N column; lanes 0-15 hold
// K=0..15 (2 per VGPR), lanes 16-31 hold K=16..31.
// Storage: row-major [Ncols x ld] where each "row" is one B-column's K run
// (e.g. kT[n][r] or vT[c][n]) => one contiguous 32B load per lane.
// ---------------------------------------------------------------------------
__device__ __forceinline__ v16h load_tileB_f16(const f16* src, int row0, int col0,
                                               int ld, int r, int hf) {
    const f16* p = src + (size_t)(row0 + r) * ld + col0 + (hf << 4);
    v8h lo = *(const v8h*)p;
    v8h hi = *(const v8h*)(p + 8);
    v16h b;
#pragma unroll
    for (int e = 0; e < 8; ++e) { b[e] = lo[e]; b[e + 8] = hi[e]; }
    return b;
}

// B-matrix 32x16 from column-major fp32 source (activation [C][N] with K=c'):
// element e of lane (r,hf) = src[(k0 + hf*16 + e)*ld + col0 + r], cvt to f16.
__device__ __forceinline__ v16h load_tileB_cm_f32(const float* src, int k0, int col0,
                                                  int ld, int r, int hf) {
    v16h b;
#pragma unroll
    for (int e = 0; e < 16; ++e) {
        b[e] = (f16)src[(size_t)(k0 + (hf << 4) + e) * ld + col0 + r];
    }
    return b;
}

// ---------------------------------------------------------------------------
// Kernel 1: convert weights fp32 -> f16 (row-major kept, contiguous in ws)
// ---------------------------------------------------------------------------
__global__ void cvt_weights_kernel(const float* __restrict__ Wq,
                                   const float* __restrict__ Wk,
                                   const float* __restrict__ Wv,
                                   f16* __restrict__ Wqh, f16* __restrict__ Wkh,
                                   f16* __restrict__ Wvh) {
    int idx = blockIdx.x * 256 + threadIdx.x;
    if (idx < RED * CH) { Wqh[idx] = (f16)Wq[idx]; Wkh[idx] = (f16)Wk[idx]; }
    if (idx < CH * CH)  { Wvh[idx] = (f16)Wv[idx]; }
}

// ---------------------------------------------------------------------------
// Kernel 2: fused Q/K/V projections via WMMA.
// All f16 weights (Wq|Wk|Wv = 160KB, contiguous in ws) are staged ONCE per
// workgroup into LDS with async global->LDS b128 copies (ASYNCcnt), then every
// A-tile fetch is a ds_load_b128.  One wave per (batch, 16-wide n-tile).
// Outputs: qT[b][n][r], kT[b][n][r] (f16), vT[b][c][n] (f16, c-major).
// ---------------------------------------------------------------------------
#define WTOT (2 * RED * CH + CH * CH)   // 81920 halves = 160 KB

__global__ __launch_bounds__(256, 1)
void proj_kernel(const float* __restrict__ skip, const float* __restrict__ gating,
                 const f16* __restrict__ Wall,   // Wq|Wk|Wv f16, contiguous
                 const float* __restrict__ bq, const float* __restrict__ bk,
                 const float* __restrict__ bv,
                 f16* __restrict__ qT, f16* __restrict__ kT, f16* __restrict__ vT) {
    __shared__ __align__(16) f16 ldsW[WTOT];    // 160 KB of the WGP's 320 KB

    const int tid  = threadIdx.x;
    const int lane = tid & 31;
    const int r    = lane & 15;
    const int hf   = lane >> 4;
    const int wid  = blockIdx.x * 8 + (tid >> 5);
    const int b    = wid >> 8;          // 256 n-tiles per batch
    const int n0   = (wid & 255) << 4;  // 16-wide n tile

    // ---- stage weights into LDS: 10240 x 16B chunks, 40 per thread --------
    for (int h0 = tid * 8; h0 < WTOT; h0 += 256 * 8) {
        async_copy_b128(ldsW + h0, Wall + h0);
    }
    wait_asynccnt0();
    __syncthreads();

    const f16* Wq_l = ldsW;
    const f16* Wk_l = ldsW + RED * CH;
    const f16* Wv_l = ldsW + 2 * RED * CH;

    const float* sk = skip   + (size_t)b * CH * NPIX;
    const float* gt = gating + (size_t)b * CH * NPIX;

    v8f accQ[2], accK[2], accV[16];
#pragma unroll
    for (int t = 0; t < 2; ++t) { accQ[t] = vzero8(); accK[t] = vzero8(); }
#pragma unroll
    for (int j = 0; j < 16; ++j) accV[j] = vzero8();

    for (int c0 = 0; c0 < CH; c0 += 32) {
        __builtin_prefetch((const void*)(sk + (size_t)(c0 + 32 + (hf << 4)) * NPIX + n0), 0, 1);
        __builtin_prefetch((const void*)(gt + (size_t)(c0 + 32 + (hf << 4)) * NPIX + n0), 0, 1);
        v16h Bs = load_tileB_cm_f32(sk, c0, n0, NPIX, r, hf);
        v16h Bg = load_tileB_cm_f32(gt, c0, n0, NPIX, r, hf);
#pragma unroll
        for (int t = 0; t < 2; ++t) {
            v16h Aq = load_tileA_f16(Wq_l, t * 16, c0, CH, r, hf);
            accQ[t] = wmma16x16x32(Aq, Bs, accQ[t]);
            v16h Ak = load_tileA_f16(Wk_l, t * 16, c0, CH, r, hf);
            accK[t] = wmma16x16x32(Ak, Bg, accK[t]);
        }
#pragma unroll
        for (int j = 0; j < 16; ++j) {
            v16h Av = load_tileA_f16(Wv_l, j * 16, c0, CH, r, hf);
            accV[j] = wmma16x16x32(Av, Bs, accV[j]);
        }
    }

    // D layout: VGPR i, lanes 0-15 -> row M=i, N=lane; lanes 16-31 -> M=i+8.
    const int ncol = n0 + r;
#pragma unroll
    for (int t = 0; t < 2; ++t) {
#pragma unroll
        for (int i = 0; i < 8; ++i) {
            const int ridx = t * 16 + i + (hf << 3);
            qT[((size_t)b * NPIX + ncol) * RED + ridx] = (f16)(accQ[t][i] + bq[ridx]);
            kT[((size_t)b * NPIX + ncol) * RED + ridx] = (f16)(accK[t][i] + bk[ridx]);
        }
    }
#pragma unroll
    for (int j = 0; j < 16; ++j) {
#pragma unroll
        for (int i = 0; i < 8; ++i) {
            const int c = j * 16 + i + (hf << 3);
            vT[((size_t)b * CH + c) * NPIX + ncol] = (f16)(accV[j][i] + bv[c]);
        }
    }
}

// ---------------------------------------------------------------------------
// Kernel 3: fused flash attention + gamma*out + skip.
// One wave per (batch, 16-row m-tile).  Per 32-col n-chunk:
//   2x S-WMMA (K-contraction R=32 in ONE wmma per tile), online softmax,
//   P staged via LDS (D-layout -> A-layout), 16x PV-WMMA into O[16x256] fp32.
// ---------------------------------------------------------------------------
__global__ __launch_bounds__(256, 1)
void attn_kernel(const float* __restrict__ skip, const float* __restrict__ gamma_p,
                 const f16* __restrict__ qT, const f16* __restrict__ kT,
                 const f16* __restrict__ vT, float* __restrict__ out) {
    __shared__ __align__(32) f16 ldsP[8 * 16 * 32];  // per-wave 16x32 P staging

    const int tid  = threadIdx.x;
    const int lane = tid & 31;
    const int r    = lane & 15;
    const int hf   = lane >> 4;
    const int w    = tid >> 5;
    f16* P = ldsP + w * (16 * 32);

    const int wid = blockIdx.x * 8 + w;
    const int b   = wid >> 8;
    const int m0  = (wid & 255) << 4;

    const f16* q = qT + (size_t)b * NPIX * RED;
    const f16* k = kT + (size_t)b * NPIX * RED;
    const f16* v = vT + (size_t)b * CH * NPIX;

    const v16h Aq = load_tileA_f16(q, m0, 0, RED, r, hf);

    v8f O[16];
#pragma unroll
    for (int j = 0; j < 16; ++j) O[j] = vzero8();
    float mst[8], lst[8];
#pragma unroll
    for (int i = 0; i < 8; ++i) { mst[i] = -__builtin_inff(); lst[i] = 0.0f; }

    const v8f cz = vzero8();

    for (int n0 = 0; n0 < NPIX; n0 += 32) {
        // stream next K rows through L2 (global_prefetch_b8)
        __builtin_prefetch((const void*)(k + (size_t)(n0 + 32 + r) * RED), 0, 1);

        v16h Bk0 = load_tileB_f16(k, n0,      0, RED, r, hf);
        v16h Bk1 = load_tileB_f16(k, n0 + 16, 0, RED, r, hf);
        v8f S0 = wmma16x16x32(Aq, Bk0, cz);
        v8f S1 = wmma16x16x32(Aq, Bk1, cz);

        float scl[8];
#pragma unroll
        for (int i = 0; i < 8; ++i) {
            // row (i + hf*8): lanes 0-15 carry rows 0-7, lanes 16-31 rows 8-15,
            // so 16-lane-group reductions (xor masks 1,2,4,8) are exact.
            float t = fmaxf(S0[i], S1[i]);
            t = fmaxf(t, __shfl_xor(t, 1, 32));
            t = fmaxf(t, __shfl_xor(t, 2, 32));
            t = fmaxf(t, __shfl_xor(t, 4, 32));
            t = fmaxf(t, __shfl_xor(t, 8, 32));
            const float mnew = fmaxf(mst[i], t);
            const float sc   = __expf(mst[i] - mnew);
            const float p0   = __expf(S0[i] - mnew);
            const float p1   = __expf(S1[i] - mnew);
            float rs = p0 + p1;
            rs += __shfl_xor(rs, 1, 32);
            rs += __shfl_xor(rs, 2, 32);
            rs += __shfl_xor(rs, 4, 32);
            rs += __shfl_xor(rs, 8, 32);
            lst[i] = lst[i] * sc + rs;
            mst[i] = mnew;
            scl[i] = sc;
            const int row = i + (hf << 3);
            P[row * 32 + r]      = (f16)p0;   // ds_store; same-wave LDS is in-order
            P[row * 32 + 16 + r] = (f16)p1;
        }

        // rescale running O by exp(m_old - m_new) per row
#pragma unroll
        for (int j = 0; j < 16; ++j) {
#pragma unroll
            for (int i = 0; i < 8; ++i) O[j][i] *= scl[i];
        }

        // P (D-layout in regs) re-read from LDS in A-layout
        const v16h Ap = load_tileA_f16(P, 0, 0, 32, r, hf);
#pragma unroll
        for (int j = 0; j < 16; ++j) {
            v16h Bv = load_tileB_f16(v, j * 16, n0, NPIX, r, hf);
            O[j] = wmma16x16x32(Ap, Bv, O[j]);
        }
    }

    const float g = gamma_p[0];
    float linv[8];
#pragma unroll
    for (int i = 0; i < 8; ++i) linv[i] = 1.0f / lst[i];

#pragma unroll
    for (int j = 0; j < 16; ++j) {
        const int c = j * 16 + r;   // O tile: N = channel, lane-indexed
#pragma unroll
        for (int i = 0; i < 8; ++i) {
            const size_t off = ((size_t)b * CH + c) * NPIX + m0 + i + (hf << 3);
            out[off] = g * O[j][i] * linv[i] + skip[off];
        }
    }
}

// ---------------------------------------------------------------------------
// Host launcher
// ---------------------------------------------------------------------------
extern "C" void kernel_launch(void* const* d_in, const int* in_sizes, int n_in,
                              void* d_out, int out_size, void* d_ws, size_t ws_size,
                              hipStream_t stream) {
    const float* skip   = (const float*)d_in[0];
    const float* gating = (const float*)d_in[1];
    const float* Wq     = (const float*)d_in[2];
    const float* bq     = (const float*)d_in[3];
    const float* Wk     = (const float*)d_in[4];
    const float* bk     = (const float*)d_in[5];
    const float* Wv     = (const float*)d_in[6];
    const float* bv     = (const float*)d_in[7];
    const float* gamma  = (const float*)d_in[8];
    float* out = (float*)d_out;

    // Workspace: qT | kT | vT | Wq(f16) | Wk(f16) | Wv(f16)  (~10.2 MB)
    f16* ws  = (f16*)d_ws;
    f16* qT  = ws;  ws += (size_t)BATCH * NPIX * RED;
    f16* kT  = ws;  ws += (size_t)BATCH * NPIX * RED;
    f16* vT  = ws;  ws += (size_t)BATCH * CH * NPIX;
    f16* Wqh = ws;  ws += (size_t)RED * CH;   // Wq|Wk|Wv contiguous
    f16* Wkh = ws;  ws += (size_t)RED * CH;
    f16* Wvh = ws;

    cvt_weights_kernel<<<(CH * CH + 255) / 256, 256, 0, stream>>>(Wq, Wk, Wv,
                                                                  Wqh, Wkh, Wvh);

    const int nblocks = (BATCH * (NPIX / 16)) / 8;  // 8 waves/block, wave32
    proj_kernel<<<nblocks, 256, 0, stream>>>(skip, gating, Wqh,
                                             bq, bk, bv, qT, kT, vT);
    attn_kernel<<<nblocks, 256, 0, stream>>>(skip, gamma, qT, kT, vT, out);
}
